// SetAbstraction_40226663694448
// MI455X (gfx1250) — matrix operand order repaired and checked by reading, not compiled, not run on hardware
//
#include <hip/hip_runtime.h>
#include <hip/hip_bf16.h>

typedef _Float16 v16h __attribute__((ext_vector_type(16)));
typedef _Float16 v8h  __attribute__((ext_vector_type(8)));
typedef float    v8f  __attribute__((ext_vector_type(8)));

// Problem constants (from reference)
static constexpr int BB = 8;
static constexpr int NN = 8192;
static constexpr int SS = 2048;
static constexpr int KK = 32;
static constexpr int CP = 61;          // points channels
static constexpr float R2 = 0.2f * 0.2f;
static constexpr long  MM = (long)BB * SS * KK;   // 524288 rows
static constexpr float EPSBN = 1e-5f;

// ---------------------------------------------------------------------------
// f32 -> f16 weight conversion
// ---------------------------------------------------------------------------
__global__ void cvt_f16_kernel(const float* __restrict__ in, _Float16* __restrict__ out, int n) {
    int i = blockIdx.x * blockDim.x + threadIdx.x;
    if (i < n) out[i] = (_Float16)in[i];
}

// ---------------------------------------------------------------------------
// Farthest point sampling: one block per batch, ONE barrier per iteration.
//  - points + running distances live in registers (8 pts/thread)
//  - coords mirrored in LDS so any thread reads centroid directly (broadcast
//    LDS read replaces owner-thread broadcast + extra barrier)
//  - 32-way cross-wave argmax done redundantly by every wave from a
//    parity-double-buffered LDS staging array (replaces second barrier;
//    single barrier/iter keeps waves <=1 iteration apart -> ping-pong safe)
// ---------------------------------------------------------------------------
__global__ __launch_bounds__(1024) void fps_kernel(const float* __restrict__ xyz,
                                                   float* __restrict__ new_xyz) {
    const int T = 1024, P = NN / 1024;      // 8 points per thread
    __shared__ float sx[NN], sy[NN], sz[NN];   // 96 KB coord mirror
    __shared__ float rv[2][32];
    __shared__ int   ri[2][32];

    int b = blockIdx.x;
    int t = threadIdx.x;
    const float* base = xyz + (size_t)b * NN * 3;

    float px[P], py[P], pz[P], dd[P];
#pragma unroll
    for (int j = 0; j < P; ++j) {
        int i = t + j * T;
        float x = base[i * 3 + 0];
        float y = base[i * 3 + 1];
        float z = base[i * 3 + 2];
        px[j] = x; py[j] = y; pz[j] = z; dd[j] = 1e10f;
        sx[i] = x; sy[i] = y; sz[i] = z;
    }
    __syncthreads();

    int lane = t & 31, wid = t >> 5;
    int far = 0;
    for (int s = 0; s < SS; ++s) {
        float cx = sx[far], cy = sy[far], cz = sz[far];
        if (t == 0) {
            float* o = new_xyz + ((size_t)b * SS + s) * 3;
            o[0] = cx; o[1] = cy; o[2] = cz;
        }
        // update distances, local argmax
        float best = -1.0f; int besti = 0;
#pragma unroll
        for (int j = 0; j < P; ++j) {
            float dx = px[j] - cx, dy = py[j] - cy, dz = pz[j] - cz;
            float d = dx * dx + dy * dy + dz * dz;
            dd[j] = fminf(dd[j], d);
            if (dd[j] > best) { best = dd[j]; besti = t + j * T; }
        }
        // wave32 argmax reduce
#pragma unroll
        for (int off = 16; off > 0; off >>= 1) {
            float ov = __shfl_down(best, off, 32);
            int   oi = __shfl_down(besti, off, 32);
            if (ov > best || (ov == best && oi < besti)) { best = ov; besti = oi; }
        }
        int p = s & 1;
        if (lane == 0) { rv[p][wid] = best; ri[p][wid] = besti; }
        __syncthreads();
        // every wave redundantly reduces the 32 wave results
        best = rv[p][lane]; besti = ri[p][lane];
#pragma unroll
        for (int off = 16; off > 0; off >>= 1) {
            float ov = __shfl_down(best, off, 32);
            int   oi = __shfl_down(besti, off, 32);
            if (ov > best || (ov == best && oi < besti)) { best = ov; besti = oi; }
        }
        far = __shfl(besti, 0, 32);
    }
}

// ---------------------------------------------------------------------------
// Ball query: one wave32 per query point; ordered append of first K indices
// within radius. cnt is wave-uniform (ballot) so the early break is uniform
// per wave; all waves still reach the single __syncthreads.
// ---------------------------------------------------------------------------
__global__ __launch_bounds__(256) void ball_query_kernel(const float* __restrict__ xyz,
                                                         const float* __restrict__ new_xyz,
                                                         int* __restrict__ idx_out) {
    int gwarp = (blockIdx.x * blockDim.x + threadIdx.x) >> 5;
    int lane  = threadIdx.x & 31;
    int wl    = threadIdx.x >> 5;
    if (gwarp >= BB * SS) return;
    int b = gwarp / SS;

    __shared__ int sidx[8][KK];

    const float* base = xyz + (size_t)b * NN * 3;
    const float* q = new_xyz + (size_t)gwarp * 3;
    float qx = q[0], qy = q[1], qz = q[2];

    int cnt = 0;
    for (int start = 0; start < NN; start += 32) {
        int i = start + lane;
        float dx = base[i * 3 + 0] - qx;
        float dy = base[i * 3 + 1] - qy;
        float dz = base[i * 3 + 2] - qz;
        float d = dx * dx + dy * dy + dz * dz;
        bool in = (d <= R2);
        unsigned long long m = __ballot(in);
        if (in) {
            int pos = cnt + __popcll(m & ((1ull << lane) - 1ull));
            if (pos < KK) sidx[wl][pos] = i;
        }
        cnt += __popcll(m);
        if (cnt >= KK) break;          // wave-uniform
    }
    __syncthreads();
    int valid = cnt < KK ? cnt : KK;
    int v = (lane < valid) ? sidx[wl][lane] : sidx[wl][0];
    idx_out[(size_t)gwarp * KK + lane] = v;
}

// ---------------------------------------------------------------------------
// Gather + concat: h0[row, c] = (c<3) ? xyz[gi]-new_xyz : points[gi, c-3]
// ---------------------------------------------------------------------------
__global__ __launch_bounds__(256) void group_kernel(const float* __restrict__ xyz,
                                                    const float* __restrict__ points,
                                                    const float* __restrict__ new_xyz,
                                                    const int* __restrict__ idx,
                                                    _Float16* __restrict__ h0) {
    long row = (long)blockIdx.x * 4 + (threadIdx.x >> 6);
    int c = threadIdx.x & 63;
    if (row >= MM) return;
    int b  = (int)(row / ((long)SS * KK));
    long rem = row % ((long)SS * KK);
    int s  = (int)(rem / KK);
    int gi = idx[row];
    float v;
    if (c < 3) {
        v = xyz[((size_t)b * NN + gi) * 3 + c] - new_xyz[((size_t)b * SS + s) * 3 + c];
    } else {
        v = points[((size_t)b * NN + gi) * CP + (c - 3)];
    }
    h0[(size_t)row * 64 + c] = (_Float16)v;
}

// ---------------------------------------------------------------------------
// Zero per-layer BN accumulators (sum at [0..128), sumsq at [128..256))
// ---------------------------------------------------------------------------
__global__ void zero_stats_kernel(float* __restrict__ stats) {
    stats[threadIdx.x] = 0.0f;
}

// ---------------------------------------------------------------------------
// WMMA GEMM: D(M x CO) = act(A)(M x 64) * W^T, accumulating per-channel
// sum / sumsq for this layer's batch norm. If FUSE, the previous layer's
// BN+ReLU (y = relu(x*a + c)) is applied on the fly to the A fragments:
// each lane's 4 groups of 8 consecutive K-channels have loop-invariant
// per-lane v8f scale/shift registers. Two chained v_wmma_f32_16x16x32_f16
// cover K = 64. W fragments loaded in-loop (L1-hot, keeps VGPRs low).
// ---------------------------------------------------------------------------
template <int NT, bool FUSE>
__global__ __launch_bounds__(256) void gemm_bn_kernel(const _Float16* __restrict__ A,
                                                      const _Float16* __restrict__ W16,
                                                      _Float16* __restrict__ D,
                                                      float* __restrict__ stats,
                                                      const float* __restrict__ prev_stats,
                                                      int nMtiles) {
    constexpr int CO = NT * 16;
    __shared__ float lsum[128];
    __shared__ float lsq[128];
    for (int i = threadIdx.x; i < 128; i += blockDim.x) { lsum[i] = 0.0f; lsq[i] = 0.0f; }
    __syncthreads();

    const int lane   = threadIdx.x & 31;
    const int wave   = threadIdx.x >> 5;
    const int row16  = lane & 15;
    const int khalfA = (lane < 16) ? 0 : 8;    // A: low lanes K0..7/16..23, high lanes +8
    const int khalfB = (lane < 16) ? 0 : 8;    // B: low lanes K0..7/16..23, high lanes +8

    v8f sc[4], sh[4];
    if constexpr (FUSE) {
#pragma unroll
        for (int g = 0; g < 4; ++g) {
            sc[g] = *(const v8f*)(prev_stats + 256 + khalfA + 16 * g);
            sh[g] = *(const v8f*)(prev_stats + 384 + khalfA + 16 * g);
        }
    }

    float ssum[NT], ssq[NT];
#pragma unroll
    for (int t = 0; t < NT; ++t) { ssum[t] = 0.0f; ssq[t] = 0.0f; }

    for (int mt = blockIdx.x * 8 + wave; mt < nMtiles; mt += gridDim.x * 8) {
        const _Float16* ap = A + ((size_t)mt * 16 + row16) * 64 + khalfA;
        v8h r0 = *(const v8h*)(ap);        // K = khalfA + 0..7
        v8h r1 = *(const v8h*)(ap + 16);   // K = khalfA + 16..23
        v8h r2 = *(const v8h*)(ap + 32);   // K = khalfA + 32..39
        v8h r3 = *(const v8h*)(ap + 48);   // K = khalfA + 48..55
        v16h a0, a1;
        if constexpr (FUSE) {
#pragma unroll
            for (int i = 0; i < 8; ++i) {
                float v0 = fmaxf((float)r0[i] * sc[0][i] + sh[0][i], 0.0f);
                float v1 = fmaxf((float)r1[i] * sc[1][i] + sh[1][i], 0.0f);
                float v2 = fmaxf((float)r2[i] * sc[2][i] + sh[2][i], 0.0f);
                float v3 = fmaxf((float)r3[i] * sc[3][i] + sh[3][i], 0.0f);
                a0[i] = (_Float16)v0; a0[8 + i] = (_Float16)v1;
                a1[i] = (_Float16)v2; a1[8 + i] = (_Float16)v3;
            }
        } else {
#pragma unroll
            for (int i = 0; i < 8; ++i) {
                a0[i] = r0[i]; a0[8 + i] = r1[i];
                a1[i] = r2[i]; a1[8 + i] = r3[i];
            }
        }
#pragma unroll
        for (int t = 0; t < NT; ++t) {
            const _Float16* wp = W16 + ((size_t)(t * 16 + row16)) * 64 + khalfB;
            v8h q0 = *(const v8h*)(wp), q1 = *(const v8h*)(wp + 16);        // K = 0..31 slice
            v8h q2 = *(const v8h*)(wp + 32), q3 = *(const v8h*)(wp + 48);   // K = 32..63 slice
            v16h b0 = __builtin_shufflevector(q0, q1, 0,1,2,3,4,5,6,7,8,9,10,11,12,13,14,15);
            v16h b1 = __builtin_shufflevector(q2, q3, 0,1,2,3,4,5,6,7,8,9,10,11,12,13,14,15);
            v8f c = {};
            c = __builtin_amdgcn_wmma_f32_16x16x32_f16(false, a0, false, b0,
                                                       (short)0, c, false, false);
            c = __builtin_amdgcn_wmma_f32_16x16x32_f16(false, a1, false, b1,
                                                       (short)0, c, false, false);
            int mrow0 = mt * 16 + ((lane < 16) ? 0 : 8);
            _Float16* dp = D + (size_t)mrow0 * CO + t * 16 + row16;
#pragma unroll
            for (int i = 0; i < 8; ++i) {
                float v = c[i];
                dp[(size_t)i * CO] = (_Float16)v;
                ssum[t] += v;
                ssq[t]  += v * v;
            }
        }
    }

    // lanes L and L+16 hold the same channel -> fold, then LDS accumulate
#pragma unroll
    for (int t = 0; t < NT; ++t) {
        float s = ssum[t] + __shfl_xor(ssum[t], 16, 32);
        float q = ssq[t]  + __shfl_xor(ssq[t], 16, 32);
        if (lane < 16) {
            atomicAdd(&lsum[t * 16 + row16], s);
            atomicAdd(&lsq[t * 16 + row16], q);
        }
    }
    __syncthreads();
    for (int i = threadIdx.x; i < CO; i += blockDim.x) {
        atomicAdd(&stats[i], lsum[i]);
        atomicAdd(&stats[128 + i], lsq[i]);
    }
}

// ---------------------------------------------------------------------------
// Fold BN stats + gamma/beta into per-channel scale/shift at [256), [384).
// (Conv bias cancels exactly under mean subtraction -> dropped.)
// ---------------------------------------------------------------------------
__global__ void finalize_stats_kernel(const float* __restrict__ g,
                                      const float* __restrict__ bt,
                                      float* __restrict__ stats, int co) {
    int c = threadIdx.x;
    if (c < co) {
        float invM = 1.0f / (float)MM;
        float mu  = stats[c] * invM;
        float var = stats[128 + c] * invM - mu * mu;
        float a = g[c] * rsqrtf(var + EPSBN);
        stats[256 + c] = a;
        stats[384 + c] = bt[c] - mu * a;
    }
}

// ---------------------------------------------------------------------------
// Final layer: BN + ReLU + max over K=32 -> new_points (f32) in d_out.
// ---------------------------------------------------------------------------
__global__ __launch_bounds__(256) void bn_apply_max_kernel(const _Float16* __restrict__ h,
                                                           const float* __restrict__ stats,
                                                           float* __restrict__ out) {
    int i = blockIdx.x * blockDim.x + threadIdx.x;   // over B*S*128
    if (i >= BB * SS * 128) return;
    int bs = i >> 7;
    int c  = i & 127;
    float a  = stats[256 + c];
    float sh = stats[384 + c];
    const _Float16* p = h + ((size_t)bs * KK) * 128 + c;
    float m = 0.0f;   // relu outputs are >= 0
#pragma unroll
    for (int k = 0; k < KK; ++k) {
        float v = (float)p[(size_t)k * 128] * a + sh;
        m = fmaxf(m, fmaxf(v, 0.0f));
    }
    out[i] = m;
}

// ---------------------------------------------------------------------------
extern "C" void kernel_launch(void* const* d_in, const int* in_sizes, int n_in,
                              void* d_out, int out_size, void* d_ws, size_t ws_size,
                              hipStream_t stream) {
    const float* xyz    = (const float*)d_in[0];
    const float* points = (const float*)d_in[1];
    const float* w0 = (const float*)d_in[2];
    const float* g0 = (const float*)d_in[4];
    const float* bt0 = (const float*)d_in[5];
    const float* w1 = (const float*)d_in[6];
    const float* g1 = (const float*)d_in[8];
    const float* bt1 = (const float*)d_in[9];
    const float* w2 = (const float*)d_in[10];
    const float* g2 = (const float*)d_in[12];
    const float* bt2 = (const float*)d_in[13];

    float* out_new_xyz    = (float*)d_out;                         // (B,S,3)
    float* out_new_points = (float*)d_out + (size_t)BB * SS * 3;   // (B,S,128)

    // workspace layout
    char* ws = (char*)d_ws;
    int*       ballidx = (int*)ws;                                   // B*S*K ints
    _Float16*  w16_0   = (_Float16*)(ws + 2097152);                  // 64x64
    _Float16*  w16_1   = (_Float16*)(ws + 2105344);                  // 64x64
    _Float16*  w16_2   = (_Float16*)(ws + 2113536);                  // 128x64
    float*     stats0  = (float*)(ws + 2129920);                     // 512 floats
    float*     stats1  = (float*)(ws + 2131968);                     // 512 floats
    float*     stats2  = (float*)(ws + 2134016);                     // 512 floats
    _Float16*  ha      = (_Float16*)(ws + 2136064);                  // M x 64 f16
    _Float16*  hb      = (_Float16*)(ws + 2136064 + (size_t)MM * 64 * 2); // M x 128 f16

    // 1. weights -> f16
    cvt_f16_kernel<<<16, 256, 0, stream>>>(w0, w16_0, 64 * 64);
    cvt_f16_kernel<<<16, 256, 0, stream>>>(w1, w16_1, 64 * 64);
    cvt_f16_kernel<<<32, 256, 0, stream>>>(w2, w16_2, 128 * 64);

    // 2. FPS -> new_xyz (d_out)
    fps_kernel<<<BB, 1024, 0, stream>>>(xyz, out_new_xyz);

    // 3. ball query
    ball_query_kernel<<<(BB * SS * 32) / 256, 256, 0, stream>>>(xyz, out_new_xyz, ballidx);

    // 4. gather/concat -> ha (M x 64 f16)
    group_kernel<<<(unsigned)(MM / 4), 256, 0, stream>>>(xyz, points, out_new_xyz, ballidx, ha);

    const int nMtiles = (int)(MM / 16);   // 32768

    // layer 0: 64 -> 64 (raw input, no fused BN)
    zero_stats_kernel<<<1, 256, 0, stream>>>(stats0);
    gemm_bn_kernel<4, false><<<512, 256, 0, stream>>>(ha, w16_0, hb, stats0, nullptr, nMtiles);
    finalize_stats_kernel<<<1, 128, 0, stream>>>(g0, bt0, stats0, 64);

    // layer 1: 64 -> 64, fusing BN0+ReLU into A-load
    zero_stats_kernel<<<1, 256, 0, stream>>>(stats1);
    gemm_bn_kernel<4, true><<<512, 256, 0, stream>>>(hb, w16_1, ha, stats1, stats0, nMtiles);
    finalize_stats_kernel<<<1, 128, 0, stream>>>(g1, bt1, stats1, 64);

    // layer 2: 64 -> 128, fusing BN1+ReLU into A-load
    zero_stats_kernel<<<1, 256, 0, stream>>>(stats2);
    gemm_bn_kernel<8, true><<<512, 256, 0, stream>>>(ha, w16_2, hb, stats2, stats1, nMtiles);
    finalize_stats_kernel<<<1, 128, 0, stream>>>(g2, bt2, stats2, 128);

    // BN2 + ReLU + max over K -> new_points
    bn_apply_max_kernel<<<(BB * SS * 128) / 256, 256, 0, stream>>>(hb, stats2, out_new_points);
}
